// GCN_58540404244885
// MI455X (gfx1250) — compile-verified, matrix operand order checked by reading, not run on hardware
//
#include <hip/hip_runtime.h>
#include <hip/hip_bf16.h>

#define F_IN  512
#define H_DIM 16
#define C_DIM 40
#define KSTEPS (F_IN / 32)          // 16 k-chunks of 32
#define PACK_ELEMS (KSTEPS * 32 * 16)   // 8192 bf16 = 16 KB

typedef __attribute__((ext_vector_type(16))) __bf16 v16bf;
typedef __attribute__((ext_vector_type(8)))  float  v8f;
typedef __attribute__((ext_vector_type(2)))  float  v2f;
typedef __attribute__((ext_vector_type(4)))  unsigned int u32x4;
typedef __attribute__((ext_vector_type(8)))  int          i32x8;
typedef __attribute__((ext_vector_type(4)))  int          i32x4;

// ---------------- utility kernels ----------------

__global__ void k_fill(float* __restrict__ p, float v, long long n) {
    long long i = blockIdx.x * (long long)blockDim.x + threadIdx.x;
    if (i < n) p[i] = v;
}

__global__ void k_deg_accum(const int* __restrict__ dst, float* __restrict__ deg, int E) {
    int i = blockIdx.x * blockDim.x + threadIdx.x;
    if (i < E) atomicAdd(&deg[dst[i]], 1.0f);
}

__global__ void k_rsqrt_inplace(float* __restrict__ p, int n) {
    int i = blockIdx.x * blockDim.x + threadIdx.x;
    if (i < n) p[i] = rsqrtf(p[i]);   // deg >= 1 (self loop), no zero guard needed
}

// Pre-pack W1[512,16] into WMMA B-fragment order: pack[(ks*32 + lane)*16 + i] holds
// bf16(W1[K, n]) with n = lane&15, kb = (lane>>4)*8, K = ks*32 + (i<8 ? kb+i : 16+kb+i-8).
__global__ void k_w1_pack(const float* __restrict__ W1, __bf16* __restrict__ pack) {
    int t = blockIdx.x * blockDim.x + threadIdx.x;
    if (t >= PACK_ELEMS) return;
    int i    = t & 15;
    int lane = (t >> 4) & 31;
    int ks   = t >> 9;
    int n  = lane & 15;
    int kb = (lane >> 4) * 8;
    int K  = ks * 32 + (i < 8 ? kb + i : 16 + kb + (i - 8));
    pack[t] = (__bf16)W1[K * H_DIM + n];
}

// ---------------- GEMM1: hlin[N,16] = x[N,512] @ W1[512,16]  (bf16 WMMA) ----------------
// One wave per 16x16 output tile; K looped in steps of 32 (16 WMMAs/tile).
// W1 B-fragments staged into LDS once per block via the Tensor Data Mover.

__global__ void __launch_bounds__(256)
k_gemm1_wmma(const float* __restrict__ x, const __bf16* __restrict__ pack,
             float* __restrict__ hlin, int tiles) {
    __shared__ alignas(32) __bf16 w1s[PACK_ELEMS];   // 16 KB

#if defined(__gfx1250__) && __has_builtin(__builtin_amdgcn_tensor_load_to_lds)
    if ((threadIdx.x >> 5) == 0) {
        // D# group0: count=1, lds_addr, global_addr[56:0], type=2 (ISA 08_async_tensor §8.3)
        unsigned long long ga = (unsigned long long)(uintptr_t)pack;
        unsigned lds_base = (unsigned)(uintptr_t)(void*)w1s;   // low 32b = wave LDS offset
        u32x4 g0 = { 1u, lds_base, (unsigned)ga,
                     (unsigned)((ga >> 32) & 0x1FFFFFFu) | (2u << 30) };
        // D# group1: 1-D copy of 4096 dwords. data_size=2(4B); tensor_dim0=4096,
        // tensor_dim1=1, tile_dim0=4096, tile_dim1=1, strides=4096. (§8.4)
        i32x8 g1 = { 0x00020000,
                     (int)(4096u << 16),          // tensor_dim0[15:0]
                     (int)(1u << 16),             // tensor_dim0[31:16]=0 | tensor_dim1[15:0]=1
                     (int)(4096u << 16),          // tensor_dim1[31:16]=0 | tile_dim0=4096
                     1,                           // tile_dim1=1, tile_dim2=0
                     4096,                        // tensor_dim0_stride[31:0]
                     (int)(4096u << 16),          // stride0[47:32]=0 | tensor_dim1_stride[15:0]
                     0 };
        i32x4 g2 = { 0, 0, 0, 0 };
        i32x4 g3 = { 0, 0, 0, 0 };
#if __clang_major__ >= 23
        i32x8 g4 = { 0, 0, 0, 0, 0, 0, 0, 0 };
        __builtin_amdgcn_tensor_load_to_lds(g0, g1, g2, g3, g4, 0);
#else
        __builtin_amdgcn_tensor_load_to_lds(g0, g1, g2, g3, 0);
#endif
        __builtin_amdgcn_s_wait_tensorcnt(0);
    }
#else
    {   // fallback: cooperative 16B-vector copy
        const uint4* src4 = (const uint4*)pack;
        uint4* dst4 = (uint4*)w1s;
        for (int i = threadIdx.x; i < (int)(PACK_ELEMS * sizeof(__bf16) / 16); i += blockDim.x)
            dst4[i] = src4[i];
    }
#endif
    __syncthreads();

    int t = threadIdx.x;
    int lane = t & 31;
    int wave = t >> 5;
    int tile = blockIdx.x * (blockDim.x >> 5) + wave;
    if (tile >= tiles) return;                       // wave-uniform

    const int mrow = tile * 16 + (lane & 15);        // A row for this lane
    const int kb   = (lane >> 4) * 8;                // K sub-offset per lane half
    const int ncol = lane & 15;                      // output column for this lane
    const float* xrow = x + (size_t)mrow * F_IN;
    const float4* xv  = (const float4*)xrow;
    const v16bf* bfrag = (const v16bf*)w1s;

    v8f acc = {};
    #pragma unroll 4
    for (int ks = 0; ks < KSTEPS; ++ks) {
        const int k0 = ks * 32;
        __builtin_prefetch(xrow + k0 + 64, 0, 1);    // global_prefetch two chunks ahead

        // A fragment: elems 0..7 -> K=k0+kb+i ; elems 8..15 -> K=k0+16+kb+i
        float af[16];
        *(float4*)&af[0]  = xv[(k0 + kb) >> 2];
        *(float4*)&af[4]  = xv[((k0 + kb) >> 2) + 1];
        *(float4*)&af[8]  = xv[(k0 + 16 + kb) >> 2];
        *(float4*)&af[12] = xv[((k0 + 16 + kb) >> 2) + 1];
        v16bf a;
        #pragma unroll
        for (int i = 0; i < 16; ++i) a[i] = (__bf16)af[i];

        // B fragment: pre-packed, one contiguous 32B LDS read per lane
        v16bf b = bfrag[ks * 32 + lane];

        acc = __builtin_amdgcn_wmma_f32_16x16x32_bf16(
                false, a, false, b, (short)0, acc, false, false);
    }

    // C/D layout: lane -> col (lane&15); VGPR r -> row r + 8*(lane>>4)
    #pragma unroll
    for (int r = 0; r < 8; ++r) {
        int m = tile * 16 + r + 8 * (lane >> 4);
        hlin[(size_t)m * H_DIM + ncol] = acc[r];
    }
}

// ---------------- edge scatter: agg[d,:] += dinv[s]*dinv[d] * h[s,:]  (16-wide) ----------------

__global__ void k_scatter16(const int* __restrict__ src, const int* __restrict__ dst,
                            const float* __restrict__ dinv, const float* __restrict__ h,
                            float* __restrict__ agg, int E, int N) {
    long long t = blockIdx.x * (long long)blockDim.x + threadIdx.x;
    int f = (int)(t & 15);
    long long e = t >> 4;
    if (e >= (long long)E + N) return;
    int s, d;
    if (e < E) { s = src[e]; d = dst[e]; }
    else       { s = d = (int)(e - E); }
    float w = dinv[s] * dinv[d];
    atomicAdd(&agg[(size_t)d * H_DIM + f], w * h[(size_t)s * H_DIM + f]);
}

// h = relu(agg + b1) into bufB, and re-zero agg (bufC) for the next aggregation pass.
__global__ void k_bias_relu_zero(float* __restrict__ agg, const float* __restrict__ b1,
                                 float* __restrict__ h, long long n16) {
    long long i = blockIdx.x * (long long)blockDim.x + threadIdx.x;
    if (i < n16) {
        float v = agg[i] + b1[i & 15];
        h[i] = v > 0.0f ? v : 0.0f;
        agg[i] = 0.0f;
    }
}

// ---------------- GEMM2: out[N,40] = agg2[N,16] @ W2[16,40] + b2  (f32 WMMA 16x16x4) ----------------

__global__ void __launch_bounds__(256)
k_gemm2_wmma(const float* __restrict__ h, const float* __restrict__ W2,
             const float* __restrict__ b2, float* __restrict__ out, int tiles) {
    int t = threadIdx.x;
    int lane = t & 31;
    int wave = t >> 5;
    int gw = blockIdx.x * (blockDim.x >> 5) + wave;
    if (gw >= tiles * 3) return;                     // wave-uniform
    int rt = gw / 3, ct = gw % 3;

    const int m    = rt * 16 + (lane & 15);
    const int ncol = ct * 16 + (lane & 15);
    const int kb   = 2 * (lane >> 4);                // f32 A 16x4: lanes<16 K={0,1}, lanes>=16 K={2,3}
    const bool nok = (ncol < C_DIM);

    v8f acc = {};
    #pragma unroll
    for (int k0 = 0; k0 < H_DIM; k0 += 4) {
        const float2 av = *(const float2*)&h[(size_t)m * H_DIM + k0 + kb];
        v2f a, b;
        a[0] = av.x;
        a[1] = av.y;
        b[0] = nok ? W2[(k0 + kb)     * C_DIM + ncol] : 0.0f;
        b[1] = nok ? W2[(k0 + kb + 1) * C_DIM + ncol] : 0.0f;
        acc = __builtin_amdgcn_wmma_f32_16x16x4_f32(
                false, a, false, b, (short)0, acc, false, false);
    }

    if (nok) {
        float bias = b2[ncol];
        #pragma unroll
        for (int r = 0; r < 8; ++r) {
            int row = rt * 16 + r + 8 * (lane >> 4);
            out[(size_t)row * C_DIM + ncol] = acc[r] + bias;
        }
    }
}

// ---------------- host launcher ----------------

extern "C" void kernel_launch(void* const* d_in, const int* in_sizes, int n_in,
                              void* d_out, int out_size, void* d_ws, size_t ws_size,
                              hipStream_t stream) {
    const float* x  = (const float*)d_in[0];
    const int*   ei = (const int*)  d_in[1];
    const float* W1 = (const float*)d_in[2];
    const float* b1 = (const float*)d_in[3];
    const float* W2 = (const float*)d_in[4];
    const float* b2 = (const float*)d_in[5];
    float* out = (float*)d_out;

    const int N = in_sizes[0] / F_IN;     // 100000 (multiple of 16)
    const int E = in_sizes[1] / 2;        // 3200000
    const int* src = ei;
    const int* dst = ei + E;

    // workspace carve-up (256B aligned): dinv | W1_pack | bufB | bufC  (~13.2 MB)
    char* ws = (char*)d_ws;
    size_t off = 0;
    auto carve = [&](size_t bytes) { void* p = ws + off; off += (bytes + 255) & ~(size_t)255; return p; };
    float*  dinv   = (float*)  carve(sizeof(float) * (size_t)N);
    __bf16* w1pack = (__bf16*) carve(sizeof(__bf16) * PACK_ELEMS);
    float*  bufB   = (float*)  carve(sizeof(float) * (size_t)N * H_DIM);  // hlin, then h
    float*  bufC   = (float*)  carve(sizeof(float) * (size_t)N * H_DIM);  // agg1, then agg2

    const long long n16 = (long long)N * H_DIM;
    const int tiles = N / 16;
    const long long scat = ((long long)E + N) * H_DIM;

    // 1) degree (init 1.0 for self loop) -> dinv = rsqrt(deg)
    k_fill<<<(N + 255) / 256, 256, 0, stream>>>(dinv, 1.0f, N);
    k_deg_accum<<<(E + 255) / 256, 256, 0, stream>>>(dst, dinv, E);
    k_rsqrt_inplace<<<(N + 255) / 256, 256, 0, stream>>>(dinv, N);

    // 2) W1 -> bf16 fragment pack ; zero agg1
    k_w1_pack<<<(PACK_ELEMS + 255) / 256, 256, 0, stream>>>(W1, w1pack);
    k_fill<<<(int)((n16 + 255) / 256), 256, 0, stream>>>(bufC, 0.0f, n16);

    // 3) hlin = x @ W1   (bf16 WMMA, TDM-staged B fragments)
    k_gemm1_wmma<<<(tiles + 7) / 8, 256, 0, stream>>>(x, w1pack, bufB, tiles);

    // 4) agg1 = Ahat @ hlin
    k_scatter16<<<(int)((scat + 255) / 256), 256, 0, stream>>>(src, dst, dinv, bufB, bufC, E, N);

    // 5) h = relu(agg1 + b1) into bufB ; agg1 -> 0 for reuse
    k_bias_relu_zero<<<(int)((n16 + 255) / 256), 256, 0, stream>>>(bufC, b1, bufB, n16);

    // 6) agg2 = Ahat @ h   ((Ahat h) W2 == Ahat (h W2))
    k_scatter16<<<(int)((scat + 255) / 256), 256, 0, stream>>>(src, dst, dinv, bufB, bufC, E, N);

    // 7) out = agg2 @ W2 + b2  (f32 WMMA)
    k_gemm2_wmma<<<(tiles * 3 + 7) / 8, 256, 0, stream>>>(bufC, W2, b2, out, tiles);
}
